// WaveletNet_14396730376550
// MI455X (gfx1250) — matrix operand order repaired and checked by reading, not compiled, not run on hardware
//
#include <hip/hip_runtime.h>
#include <cstdint>
#include <cstddef>

typedef __bf16 bf16_t;
typedef __attribute__((ext_vector_type(16))) __bf16 v16bf;
typedef __attribute__((ext_vector_type(8)))  __bf16 v8bf;
typedef __attribute__((ext_vector_type(8)))  float  v8f;
typedef __attribute__((ext_vector_type(4)))  float  v4f;
typedef __attribute__((ext_vector_type(4)))  int    v4i;

#define INV_SQRT2F 0.70710678118654752440f
static constexpr int OUT_N = 10;

// ---------- CDNA5 async global->LDS path (guarded; falls back to sync copy) ----------
#if defined(__gfx1250__) && __has_builtin(__builtin_amdgcn_global_load_async_to_lds_b128)
#define USE_ASYNC_LDS 1
typedef __attribute__((address_space(1))) v4i as1_v4i;
typedef __attribute__((address_space(3))) v4i as3_v4i;
#endif

__device__ __forceinline__ void cp16(const bf16_t* g, bf16_t* l) {
#ifdef USE_ASYNC_LDS
  __builtin_amdgcn_global_load_async_to_lds_b128(
      (as1_v4i*)(uintptr_t)(const void*)g, (as3_v4i*)l, 0, 0);
#else
  *(v8bf*)l = *(const v8bf*)g;
#endif
}

// Wait until at most N async global->LDS ops remain in flight (per-wave ASYNCcnt).
template <int N>
__device__ __forceinline__ void stage_wait_n() {
#ifdef USE_ASYNC_LDS
#if __has_builtin(__builtin_amdgcn_s_wait_asynccnt)
  __builtin_amdgcn_s_wait_asynccnt(N);
#else
  asm volatile("s_wait_asynccnt %0" :: "n"(N) : "memory");
#endif
#endif
}

// ---------- WMMA helpers ----------
__device__ __forceinline__ v16bf cat8(v8bf lo, v8bf hi) {
  return __builtin_shufflevector(lo, hi, 0,1,2,3,4,5,6,7,8,9,10,11,12,13,14,15);
}

// A fragment (16x32 bf16): lane = M (l&15); K = (l>>4)*8 + e (e<8), 16 + (l>>4)*8 + (e-8) (e>=8)
__device__ __forceinline__ v16bf frag_a(const bf16_t* s, int mrow, int kk, int lane) {
  const bf16_t* p = s + (mrow + (lane & 15)) * 72 + kk + ((lane >> 4) << 3);
  return cat8(*(const v8bf*)p, *(const v8bf*)(p + 16));
}
// B fragment (32x16 bf16), LDS holds W transposed [N][K]: lane = N (l&15); K = (l>>4)*8 + e (e<8), 16 + (l>>4)*8 + (e-8) (e>=8)
__device__ __forceinline__ v16bf frag_b(const bf16_t* s, int nrow, int kk, int lane) {
  const bf16_t* p = s + (nrow + (lane & 15)) * 72 + kk + ((lane >> 4) << 3);
  return cat8(*(const v8bf*)p, *(const v8bf*)(p + 16));
}

__device__ __forceinline__ v8f wmma_bf16(v16bf a, v16bf b, v8f c) {
#if defined(__gfx1250__)
  return __builtin_amdgcn_wmma_f32_16x16x32_bf16(false, a, false, b, (short)0, c, false, false);
#else
  return c;
#endif
}

// ---------- weight prep kernels (fold Haar DWT into W1; transpose to [N][K] bf16) ----------
__global__ void prep_w1_kernel(const float* __restrict__ W1, bf16_t* __restrict__ W1t) {
  const int idx = blockIdx.x * 256 + threadIdx.x;   // 2048 blocks
  const int n = idx & 255;
  const int j = idx >> 8;                           // 0..2047
  const float a = W1[(size_t)j * 256 + n];
  const float d = W1[(size_t)(j + 2048) * 256 + n];
  W1t[(size_t)n * 4096 + 2 * j]     = (bf16_t)((a + d) * INV_SQRT2F);
  W1t[(size_t)n * 4096 + 2 * j + 1] = (bf16_t)((a - d) * INV_SQRT2F);
}

__global__ void prep_w2_kernel(const float* __restrict__ W2, bf16_t* __restrict__ W2t) {
  const int idx = blockIdx.x * 256 + threadIdx.x;   // 512 blocks
  const int n = idx & 511;
  const int k = idx >> 9;                           // 0..255
  W2t[(size_t)n * 256 + k] = (bf16_t)W2[(size_t)k * 512 + n];
}

__global__ void prep_w3_kernel(const float* __restrict__ W3, bf16_t* __restrict__ W3t) {
  const int idx = blockIdx.x * 256 + threadIdx.x;   // 32 blocks
  const int n = idx & 15;
  const int k = idx >> 4;                           // 0..511
  W3t[(size_t)n * 512 + k] = (n < OUT_N) ? (bf16_t)W3[(size_t)k * OUT_N + n] : (bf16_t)0.0f;
}

// ---------- main tiled GEMM: C = relu(A @ Bt^T + bias), bf16 out ----------
// 256 threads (8 waves). tile 128(M) x 64(N), BK=64, double-buffered LDS with async DMA
// prefetch: next tile's global->LDS DMA runs underneath the current tile's WMMAs.
template <bool A_F32>
__global__ __launch_bounds__(256) void gemm_bf16_relu_kernel(
    const void* __restrict__ Aptr, const bf16_t* __restrict__ Bt,
    const float* __restrict__ bias, bf16_t* __restrict__ Out,
    int K, int N)
{
  __shared__ __align__(16) bf16_t sA[2][128 * 72];
  __shared__ __align__(16) bf16_t sB[2][64 * 72];

  const int t    = threadIdx.x;
  const int lane = t & 31;
  const int wave = t >> 5;
  const int wm   = (wave & 3) * 32;
  const int wn   = (wave >> 2) * 32;
  const int m0   = blockIdx.x * 128;
  const int n0   = blockIdx.y * 64;

  v8f acc00 = {0,0,0,0,0,0,0,0};
  v8f acc01 = acc00, acc10 = acc00, acc11 = acc00;

  const int arow = t >> 1, ah = (t & 1) * 32;       // A stage: 128 rows x 2 halves
  const int brow = t >> 2, bc = (t & 3) * 16;       // B stage: 64 rows x 4 chunks

  // per-wave async instructions issued per stage (for partial ASYNCcnt waits)
  constexpr int PEND = A_F32 ? 2 : 6;

  auto stage = [&](int buf, int k0) {
    { // 64x64 bf16 weight tile, [N][K] transposed layout -> async DMA to LDS
      const bf16_t* g = Bt + (size_t)(n0 + brow) * K + k0 + bc;
      bf16_t* l = sB[buf] + brow * 72 + bc;
      cp16(g, l);
      cp16(g + 8, l + 8);
    }
    if constexpr (A_F32) { // 128x64 f32 activations: non-temporal load, cvt->bf16, LDS store
      const v4f* g = (const v4f*)((const float*)Aptr + (size_t)(m0 + arow) * K + k0 + ah);
      bf16_t* l = sA[buf] + arow * 72 + ah;
      #pragma unroll
      for (int i = 0; i < 4; ++i) {
        v4f f0 = __builtin_nontemporal_load(g + 2 * i);
        v4f f1 = __builtin_nontemporal_load(g + 2 * i + 1);
        v8bf v;
        v[0] = (bf16_t)f0[0]; v[1] = (bf16_t)f0[1]; v[2] = (bf16_t)f0[2]; v[3] = (bf16_t)f0[3];
        v[4] = (bf16_t)f1[0]; v[5] = (bf16_t)f1[1]; v[6] = (bf16_t)f1[2]; v[7] = (bf16_t)f1[3];
        *(v8bf*)(l + i * 8) = v;
      }
    } else {               // 128x64 bf16 activations -> async DMA to LDS
      const bf16_t* g = (const bf16_t*)Aptr + (size_t)(m0 + arow) * K + k0 + ah;
      bf16_t* l = sA[buf] + arow * 72 + ah;
      #pragma unroll
      for (int i = 0; i < 4; ++i) cp16(g + i * 8, l + i * 8);
    }
  };

  const int T = K >> 6;   // number of BK=64 tiles
  stage(0, 0);
  for (int i = 0; i < T; ++i) {
    const int cur = i & 1;
    if (i + 1 < T) {
      stage(cur ^ 1, (i + 1) << 6);   // prefetch next tile (DMA overlaps compute)
      stage_wait_n<PEND>();           // tile i complete; tile i+1 still in flight
    } else {
      stage_wait_n<0>();
    }
    __syncthreads();

    const bf16_t* cA = sA[cur];
    const bf16_t* cB = sB[cur];
    #pragma unroll
    for (int kk = 0; kk < 64; kk += 32) {
      v16bf a0 = frag_a(cA, wm,      kk, lane);
      v16bf a1 = frag_a(cA, wm + 16, kk, lane);
      v16bf b0 = frag_b(cB, wn,      kk, lane);
      v16bf b1 = frag_b(cB, wn + 16, kk, lane);
      acc00 = wmma_bf16(a0, b0, acc00);
      acc01 = wmma_bf16(a0, b1, acc01);
      acc10 = wmma_bf16(a1, b0, acc10);
      acc11 = wmma_bf16(a1, b1, acc11);
    }
    __syncthreads();  // all waves done reading buf[cur] before it is restaged
  }

  // epilogue: bias + relu, bf16 store. C layout: VGPR r -> M=r (lanes 0-15) / r+8 (16-31), N=lane&15
  const int lcol = lane & 15, lhi = (lane >> 4) * 8;
  v8f accs[2][2] = {{acc00, acc01}, {acc10, acc11}};
  #pragma unroll
  for (int mi = 0; mi < 2; ++mi) {
    #pragma unroll
    for (int ni = 0; ni < 2; ++ni) {
      const int col = n0 + wn + ni * 16 + lcol;
      const float bv = bias[col];
      #pragma unroll
      for (int r = 0; r < 8; ++r) {
        const int row = m0 + wm + mi * 16 + r + lhi;
        float v = accs[mi][ni][r] + bv;
        v = v > 0.0f ? v : 0.0f;
        Out[(size_t)row * N + col] = (bf16_t)v;
      }
    }
  }
}

// ---------- final layer: out[16384,10] = h2[16384,512] @ W3t[16,512]^T + b3 (f32) ----------
__global__ __launch_bounds__(256) void gemm3_kernel(
    const bf16_t* __restrict__ H2, const bf16_t* __restrict__ W3t,
    const float* __restrict__ b3, float* __restrict__ Out)
{
  __shared__ __align__(16) bf16_t sB[16 * 520];
  const int t = threadIdx.x;
  { // preload all of W3t (16 x 512) once via async DMA
    const int row = t >> 4, seg = (t & 15) * 32;
    const bf16_t* g = W3t + row * 512 + seg;
    bf16_t* l = sB + row * 520 + seg;
    #pragma unroll
    for (int i = 0; i < 4; ++i) cp16(g + i * 8, l + i * 8);
  }
  stage_wait_n<0>();
  __syncthreads();

  const int lane = t & 31, wave = t >> 5;
  const int m0 = blockIdx.x * 128 + wave * 16;
  const int lrow = lane & 15, lhalf = lane >> 4;
  const bf16_t* arow = H2 + (size_t)(m0 + lrow) * 512;

  v8f acc = {0,0,0,0,0,0,0,0};
  #pragma unroll 4
  for (int k = 0; k < 512; k += 32) {
    const bf16_t* p = arow + k + (lhalf << 3);
    v16bf a = cat8(*(const v8bf*)p, *(const v8bf*)(p + 16));
    const bf16_t* q = sB + lrow * 520 + k + (lhalf << 3);
    v16bf b = cat8(*(const v8bf*)q, *(const v8bf*)(q + 16));
    acc = wmma_bf16(a, b, acc);
  }

  if (lrow < OUT_N) {
    const float bv = b3[lrow];
    #pragma unroll
    for (int r = 0; r < 8; ++r) {
      const int row = m0 + r + lhalf * 8;
      Out[(size_t)row * OUT_N + lrow] = acc[r] + bv;
    }
  }
}

extern "C" void kernel_launch(void* const* d_in, const int* in_sizes, int n_in,
                              void* d_out, int out_size, void* d_ws, size_t ws_size,
                              hipStream_t stream) {
  const float* x  = (const float*)d_in[0];
  const float* W1 = (const float*)d_in[1];
  const float* b1 = (const float*)d_in[2];
  const float* W2 = (const float*)d_in[3];
  const float* b2 = (const float*)d_in[4];
  const float* W3 = (const float*)d_in[5];
  const float* b3 = (const float*)d_in[6];
  float* out = (float*)d_out;

  char* ws = (char*)d_ws;
  bf16_t* W1t = (bf16_t*)(ws);                 // 256*4096*2  = 2 MB
  bf16_t* W2t = (bf16_t*)(ws + 2097152);       // 512*256*2   = 256 KB
  bf16_t* W3t = (bf16_t*)(ws + 2359296);       // 16*512*2    = 16 KB
  bf16_t* h1  = (bf16_t*)(ws + 2375680);       // 16384*256*2 = 8 MB
  bf16_t* h2  = (bf16_t*)(ws + 10764288);      // 16384*512*2 = 16 MB

  prep_w1_kernel<<<2048, 256, 0, stream>>>(W1, W1t);
  prep_w2_kernel<<<512, 256, 0, stream>>>(W2, W2t);
  prep_w3_kernel<<<32, 256, 0, stream>>>(W3, W3t);

  // GEMM1: relu(x @ (D*W1) + b1) -> h1 [16384,256]  (DWT folded into weights)
  gemm_bf16_relu_kernel<true><<<dim3(128, 4), 256, 0, stream>>>(x, W1t, b1, h1, 4096, 256);
  // GEMM2: relu(h1 @ W2 + b2) -> h2 [16384,512]
  gemm_bf16_relu_kernel<false><<<dim3(128, 8), 256, 0, stream>>>(h1, W2t, b2, h2, 256, 512);
  // GEMM3: h2 @ W3 + b3 -> out [16384,10]
  gemm3_kernel<<<128, 256, 0, stream>>>(h2, W3t, b3, out);

  (void)in_sizes; (void)n_in; (void)out_size; (void)ws_size;
}
